// RWKV6Attention_89352499626495
// MI455X (gfx1250) — matrix operand order repaired and checked by reading, not compiled, not run on hardware
//
#include <hip/hip_runtime.h>

// ---------------------------------------------------------------------------
// RWKV6 forward for MI455X (gfx1250, wave32, WMMA bf16 16x16x32, async LDS copy)
// ---------------------------------------------------------------------------

typedef __bf16 bf16;
typedef __attribute__((ext_vector_type(16))) __bf16 v16bf;
typedef __attribute__((ext_vector_type(8)))  __bf16 v8bf;
typedef __attribute__((ext_vector_type(8)))  float  v8f;
typedef __attribute__((ext_vector_type(4)))  int    v4i;

static constexpr int Bsz    = 2;
static constexpr int Tlen   = 4096;
static constexpr int Dm     = 2048;
static constexpr int NH     = 32;
static constexpr int HDm    = 64;
static constexpr int CH     = 32;          // chunk length
static constexpr int NCHUNK = Tlen / CH;   // 128
static constexpr int BT     = Bsz * Tlen;  // 8192 rows
static constexpr int MIXC   = 32;
static constexpr int DECC   = 64;

#define WMMA_BF16(a, b, c) \
  __builtin_amdgcn_wmma_f32_16x16x32_bf16(false, (a), false, (b), (short)0, (c), false, false)

// ---- async global -> LDS copy (CDNA5 GLOBAL_LOAD_ASYNC_TO_LDS_B128) --------
#if defined(__gfx1250__) && __has_builtin(__builtin_amdgcn_global_load_async_to_lds_b128)
#define HAVE_ASYNC_LDS 1
#else
#define HAVE_ASYNC_LDS 0
#endif

__device__ inline void cp16_async(void* lds, const void* g) {
#if HAVE_ASYNC_LDS
  __builtin_amdgcn_global_load_async_to_lds_b128(
      (__attribute__((address_space(1))) v4i*)g,
      (__attribute__((address_space(3))) v4i*)lds, 0, 0);
#else
  *(v8bf*)lds = *(const v8bf*)g;
#endif
}

__device__ inline void async_wait0() {
#if HAVE_ASYNC_LDS
#if __has_builtin(__builtin_amdgcn_s_wait_asynccnt)
  __builtin_amdgcn_s_wait_asynccnt(0);
#else
  asm volatile("s_wait_asynccnt 0x0" ::: "memory");
#endif
#endif
}

// ---- WMMA fragment builders (ISA 7.12.2 layouts, wave32) -------------------
// A-matrix 16x32 bf16: row = lane%16; K chunks {c0..c0+7, 16+c0..16+c0+7},
// c0 = (lane>=16)*8.  Two 16-byte LDS loads per lane.
__device__ inline v16bf frag_a(const bf16* base, int stride, int lane, int kbase) {
  const int row = lane & 15;
  const int c0  = (lane >> 4) << 3;
  const bf16* p = base + (size_t)row * stride + kbase;
  union { v16bf v; v8bf h[2]; } u;
  u.h[0] = *(const v8bf*)(p + c0);
  u.h[1] = *(const v8bf*)(p + 16 + c0);
  return u.v;
}
// B-matrix 32x16 bf16: col n = lane%16; K chunks {c0..c0+7, 16+c0..16+c0+7},
// c0 = (lane>=16)*8.  Two 16-byte LDS loads per lane.
__device__ inline v16bf frag_b(const bf16* base, int stride, int lane, int kbase) {
  const int n    = lane & 15;
  const int c0   = (lane >> 4) << 3;
  const bf16* p  = base + (size_t)n * stride + kbase;
  union { v16bf v; v8bf h[2]; } u;
  u.h[0] = *(const v8bf*)(p + c0);
  u.h[1] = *(const v8bf*)(p + 16 + c0);
  return u.v;
}

// ---------------------------------------------------------------------------
// Conversions
// ---------------------------------------------------------------------------
__global__ void cvt_f2b_kernel(const float* __restrict__ in, bf16* __restrict__ out, int n) {
  int i = blockIdx.x * blockDim.x + threadIdx.x;
  if (i < n) out[i] = (bf16)in[i];
}
// decay_w1 (Dm x DECC) -> transposed bf16 (DECC x Dm)
__global__ void cvt_dw1T_kernel(const float* __restrict__ in, bf16* __restrict__ out) {
  int i = blockIdx.x * blockDim.x + threadIdx.x;
  if (i >= Dm * DECC) return;
  int d = i / DECC, j = i % DECC;
  out[(size_t)j * Dm + d] = (bf16)in[i];
}
// decay_w2 (DECC x Dm) -> transposed bf16 (Dm x DECC)
__global__ void cvt_dw2T_kernel(const float* __restrict__ in, bf16* __restrict__ out) {
  int i = blockIdx.x * blockDim.x + threadIdx.x;
  if (i >= Dm * DECC) return;
  int j = i / Dm, d = i % Dm;
  out[(size_t)d * DECC + j] = (bf16)in[i];
}

// ---------------------------------------------------------------------------
// Token-shift + data-dependent mixing -> xr/xk/xv/xw/xg (bf16)
// ---------------------------------------------------------------------------
__global__ __launch_bounds__(256) void mix_kernel(
    const float* __restrict__ x,
    const float* __restrict__ maa_x, const float* __restrict__ maa_r,
    const float* __restrict__ maa_k, const float* __restrict__ maa_v,
    const float* __restrict__ maa_w, const float* __restrict__ maa_g,
    const float* __restrict__ w1,   // Dm x 5*MIXC
    const float* __restrict__ w2,   // 5*MIXC x Dm
    bf16* __restrict__ oxr, bf16* __restrict__ oxk, bf16* __restrict__ oxv,
    bf16* __restrict__ oxw, bf16* __restrict__ oxg) {
  const int row = blockIdx.x;            // 0..BT-1
  const int t   = row & (Tlen - 1);
  const int tid = threadIdx.x;
  __shared__ float sxx[Dm];
  __shared__ float sdx[Dm];
  __shared__ float st5[5 * MIXC];
  const float* xrow = x + (size_t)row * Dm;

  for (int d = tid; d < Dm; d += 256) {
    float xv0 = xrow[d];
    float xp  = (t == 0) ? 0.f : xrow[d - Dm];
    float dx  = xp - xv0;
    sdx[d] = dx;
    sxx[d] = xv0 + dx * maa_x[d];
  }
  __syncthreads();
  if (tid < 5 * MIXC) {
    float acc = 0.f;
    for (int d = 0; d < Dm; ++d) acc += sxx[d] * w1[(size_t)d * (5 * MIXC) + tid];
    st5[tid] = tanhf(acc);
  }
  __syncthreads();
  for (int d = tid; d < Dm; d += 256) {
    float xv0 = xrow[d], dx = sdx[d];
    float m0 = 0.f, m1 = 0.f, m2 = 0.f, m3 = 0.f, m4 = 0.f;
#pragma unroll
    for (int r = 0; r < MIXC; ++r) {
      m0 += st5[0 * MIXC + r] * w2[(size_t)(0 * MIXC + r) * Dm + d];
      m1 += st5[1 * MIXC + r] * w2[(size_t)(1 * MIXC + r) * Dm + d];
      m2 += st5[2 * MIXC + r] * w2[(size_t)(2 * MIXC + r) * Dm + d];
      m3 += st5[3 * MIXC + r] * w2[(size_t)(3 * MIXC + r) * Dm + d];
      m4 += st5[4 * MIXC + r] * w2[(size_t)(4 * MIXC + r) * Dm + d];
    }
    size_t o = (size_t)row * Dm + d;
    oxr[o] = (bf16)(xv0 + dx * (maa_r[d] + m0));
    oxk[o] = (bf16)(xv0 + dx * (maa_k[d] + m1));
    oxv[o] = (bf16)(xv0 + dx * (maa_v[d] + m2));
    oxw[o] = (bf16)(xv0 + dx * (maa_w[d] + m3));
    oxg[o] = (bf16)(xv0 + dx * (maa_g[d] + m4));
  }
}

// ---------------------------------------------------------------------------
// Generic WMMA GEMM:  C[m,n] = sum_k A[m,k] * W[n,k]   (A: MxK, W: NxK, bf16)
// BM=128 BN=64 BK=64; 8 waves, each a 32x32 tile; double-buffered async LDS.
// ---------------------------------------------------------------------------
__global__ __launch_bounds__(256) void gemm_bf16_kernel(
    const bf16* __restrict__ A, const bf16* __restrict__ W,
    float* __restrict__ Cout, int M, int N, int K) {
  __shared__ bf16 As[2][128][72];
  __shared__ bf16 Ws[2][64][72];
  const int tid = threadIdx.x, lane = tid & 31, wave = tid >> 5;
  const int bm = blockIdx.y * 128, bn = blockIdx.x * 64;
  const int wm = (wave >> 1) * 32, wn = (wave & 1) * 32;
  v8f acc[2][2] = {};
  const int nstage = K >> 6;

  auto issue = [&](int buf, int k0) {
#pragma unroll
    for (int i = 0; i < 4; ++i) {            // A: 128 rows x 8 chunks of 8 bf16
      int c = tid + i * 256;                 // 0..1023
      int row = c >> 3, col = (c & 7) << 3;
      cp16_async(&As[buf][row][col], A + (size_t)(bm + row) * K + k0 + col);
    }
#pragma unroll
    for (int i = 0; i < 2; ++i) {            // W: 64 rows x 8 chunks of 8 bf16
      int c = tid + i * 256;                 // 0..511
      int row = c >> 3, col = (c & 7) << 3;
      cp16_async(&Ws[buf][row][col], W + (size_t)(bn + row) * K + k0 + col);
    }
  };

  issue(0, 0);
  for (int s = 0; s < nstage; ++s) {
    async_wait0();
    __syncthreads();                          // stage s tile visible to all waves
    if (s + 1 < nstage) issue((s + 1) & 1, (s + 1) << 6);  // overlap with compute
    const int cur = s & 1;
#pragma unroll
    for (int kk = 0; kk < 64; kk += 32)
#pragma unroll
      for (int mt = 0; mt < 2; ++mt)
#pragma unroll
        for (int nt = 0; nt < 2; ++nt) {
          v16bf af  = frag_a(&As[cur][wm + mt * 16][0], 72, lane, kk);
          v16bf bfv = frag_b(&Ws[cur][wn + nt * 16][0], 72, lane, kk);
          acc[mt][nt] = WMMA_BF16(af, bfv, acc[mt][nt]);
        }
  }
#pragma unroll
  for (int mt = 0; mt < 2; ++mt)
#pragma unroll
    for (int nt = 0; nt < 2; ++nt)
#pragma unroll
      for (int i = 0; i < 8; ++i) {
        int row = bm + wm + mt * 16 + ((lane >> 4) << 3) + i;
        int col = bn + wn + nt * 16 + (lane & 15);
        Cout[(size_t)row * N + col] = acc[mt][nt][i];
      }
}

// ---------------------------------------------------------------------------
// Small elementwise kernels
// ---------------------------------------------------------------------------
__global__ void tanh_cvt_kernel(const float* __restrict__ in, bf16* __restrict__ out, int n) {
  int i = blockIdx.x * blockDim.x + threadIdx.x;
  if (i < n) out[i] = (bf16)tanhf(in[i]);
}

// wdelta -> logw (in place); k *= 1-exp(logw); gpre -> silu(gpre+bg) (in place)
__global__ void prep_kernel(float* __restrict__ wdelta, float* __restrict__ kbuf,
                            float* __restrict__ gbuf,
                            const float* __restrict__ time_decay,
                            const float* __restrict__ bg, int total) {
  int i = blockIdx.x * blockDim.x + threadIdx.x;
  if (i >= total) return;
  int d = i & (Dm - 1);
  float w  = time_decay[d] + wdelta[i];
  float lw = -__expf(w);
  wdelta[i] = lw;
  kbuf[i]  *= (1.f - __expf(lw));
  float gp = gbuf[i] + bg[d];
  gbuf[i] = gp / (1.f + __expf(-gp));
}

__global__ void mulg_kernel(const float* __restrict__ y, const float* __restrict__ g,
                            bf16* __restrict__ out, int n) {
  int i = blockIdx.x * blockDim.x + threadIdx.x;
  if (i < n) out[i] = (bf16)(y[i] * g[i]);
}

// ---------------------------------------------------------------------------
// Chunked linear attention. One block (4 waves) per (b,h); state S (64x64)
// lives in LDS (fp32 master + bf16 transposed copy) across all 128 chunks.
// Next-chunk q/k/v/logw prefetched (global_prefetch) during current compute.
// ---------------------------------------------------------------------------
__global__ __launch_bounds__(128) void rwkv_attn_kernel(
    const float* __restrict__ qf, const float* __restrict__ kf,
    const float* __restrict__ vf, const float* __restrict__ lwf,
    float* __restrict__ y) {
  const int bh = blockIdx.x;
  const int b = bh >> 5;          // NH = 32
  const int h = bh & 31;
  const int tid = threadIdx.x, lane = tid & 31, wave = tid >> 5;

  __shared__ float Ll[CH][HDm];   // logw, then in-place cumsum L
  __shared__ float Lc[HDm];       // chunk-total decay
  __shared__ bf16  qd[CH][72];    // q * exp(L)
  __shared__ bf16  kd[CH][72];    // k * exp(-L)
  __shared__ bf16  keT[HDm][40];  // kend transposed: [hd][c]
  __shared__ bf16  vT[HDm][40];   // v transposed:    [hd][c]
  __shared__ bf16  Am[CH][40];    // masked intra-chunk A (bf16)
  __shared__ float Sf[HDm][72];   // state S[k][v], fp32 master
  __shared__ bf16  STb[HDm][72];  // S transposed bf16: [v][k] (B-fragment layout)

  for (int i = tid; i < HDm * 72; i += 128) {
    (&Sf[0][0])[i]  = 0.f;
    (&STb[0][0])[i] = (bf16)0.f;
  }
  __syncthreads();

  for (int n = 0; n < NCHUNK; ++n) {
    const int t0 = n * CH;
    // ---- load logw chunk ----
    for (int i = tid; i < CH * HDm; i += 128) {
      int c = i >> 6, hd = i & 63;
      Ll[c][hd] = lwf[((size_t)(b * Tlen + t0 + c)) * Dm + h * HDm + hd];
    }
    __syncthreads();
    // ---- cumsum along chunk ----
    if (tid < HDm) {
      float s = 0.f;
      for (int c = 0; c < CH; ++c) { s += Ll[c][tid]; Ll[c][tid] = s; }
      Lc[tid] = s;
    }
    __syncthreads();
    // ---- decay-weighted operands (+ prefetch next chunk into cache) ----
    for (int i = tid; i < CH * HDm; i += 128) {
      int c = i >> 6, hd = i & 63;
      size_t e = ((size_t)(b * Tlen + t0 + c)) * Dm + h * HDm + hd;
      float L  = Ll[c][hd];
      float q_ = qf[e], k_ = kf[e], v_ = vf[e];
      qd[c][hd]  = (bf16)(q_ * __expf(L));
      kd[c][hd]  = (bf16)(k_ * __expf(-L));
      keT[hd][c] = (bf16)(k_ * __expf(Lc[hd] - L));
      vT[hd][c]  = (bf16)v_;
      if (n + 1 < NCHUNK && (i & 1024) == 0 && (hd & 31) == 0) {
        size_t en = e + (size_t)CH * Dm;    // one prefetch per 128B line
        __builtin_prefetch(&qf[en], 0, 0);
        __builtin_prefetch(&kf[en], 0, 0);
        __builtin_prefetch(&vf[en], 0, 0);
        __builtin_prefetch(&lwf[en], 0, 0);
      }
    }
    __syncthreads();
    // ---- A = qd @ kd^T (32x32), tril mask, -> bf16 ----
    {
      const int mt = wave >> 1, nt = wave & 1;
      v8f a = {};
#pragma unroll
      for (int ks = 0; ks < HDm; ks += 32) {
        v16bf af  = frag_a(&qd[mt * 16][0], 72, lane, ks);
        v16bf bfv = frag_b(&kd[nt * 16][0], 72, lane, ks);
        a = WMMA_BF16(af, bfv, a);
      }
#pragma unroll
      for (int i = 0; i < 8; ++i) {
        int row = mt * 16 + ((lane >> 4) << 3) + i;
        int col = nt * 16 + (lane & 15);
        Am[row][col] = (col <= row) ? (bf16)a[i] : (bf16)0.f;
      }
    }
    __syncthreads();
    // ---- y = Am @ v + qd @ S  (32x64) ----
    {
      const int mt = wave >> 1;
      const int ntb = (wave & 1) * 2;
#pragma unroll
      for (int j = 0; j < 2; ++j) {
        const int nt = ntb + j;
        v8f a = {};
#pragma unroll
        for (int ks = 0; ks < HDm; ks += 32) {
          v16bf af  = frag_a(&qd[mt * 16][0], 72, lane, ks);
          v16bf bfv = frag_b(&STb[nt * 16][0], 72, lane, ks);
          a = WMMA_BF16(af, bfv, a);
        }
        v16bf af2 = frag_a(&Am[mt * 16][0], 40, lane, 0);
        v16bf bf2 = frag_b(&vT[nt * 16][0], 40, lane, 0);
        a = WMMA_BF16(af2, bf2, a);
#pragma unroll
        for (int i = 0; i < 8; ++i) {
          int row = mt * 16 + ((lane >> 4) << 3) + i;
          int col = nt * 16 + (lane & 15);
          y[((size_t)(b * Tlen + t0 + row)) * Dm + h * HDm + col] = a[i];
        }
      }
    }
    __syncthreads();
    // ---- S = diag(exp(Lc)) * S + kend^T @ v  (64x64), seeded via C operand ----
    {
      const int m0 = wave * 16;
#pragma unroll
      for (int nt = 0; nt < 4; ++nt) {
        v8f cin;
#pragma unroll
        for (int i = 0; i < 8; ++i) {
          int row = m0 + ((lane >> 4) << 3) + i;
          int col = nt * 16 + (lane & 15);
          cin[i] = Sf[row][col] * __expf(Lc[row]);
        }
        v16bf af  = frag_a(&keT[m0][0], 40, lane, 0);
        v16bf bfv = frag_b(&vT[nt * 16][0], 40, lane, 0);
        v8f sv = WMMA_BF16(af, bfv, cin);
#pragma unroll
        for (int i = 0; i < 8; ++i) {
          int row = m0 + ((lane >> 4) << 3) + i;
          int col = nt * 16 + (lane & 15);
          Sf[row][col]  = sv[i];
          STb[col][row] = (bf16)sv[i];
        }
      }
    }
    __syncthreads();
  }
}

// ---------------------------------------------------------------------------
// Host launcher
// ---------------------------------------------------------------------------
extern "C" void kernel_launch(void* const* d_in, const int* in_sizes, int n_in,
                              void* d_out, int out_size, void* d_ws, size_t ws_size,
                              hipStream_t stream) {
  (void)in_sizes; (void)n_in; (void)out_size; (void)ws_size;

  const float* x          = (const float*)d_in[0];
  const float* Wq         = (const float*)d_in[1];
  const float* Wk         = (const float*)d_in[2];
  const float* Wv         = (const float*)d_in[3];
  const float* Wo         = (const float*)d_in[4];
  const float* Wg         = (const float*)d_in[5];
  const float* bg         = (const float*)d_in[6];
  const float* maa_x      = (const float*)d_in[7];
  const float* maa_r      = (const float*)d_in[8];
  const float* maa_k      = (const float*)d_in[9];
  const float* maa_v      = (const float*)d_in[10];
  const float* maa_w      = (const float*)d_in[11];
  const float* maa_g      = (const float*)d_in[12];
  const float* maa_w1     = (const float*)d_in[13];
  const float* maa_w2     = (const float*)d_in[14];
  const float* time_decay = (const float*)d_in[15];
  const float* decay_w1   = (const float*)d_in[16];
  const float* decay_w2   = (const float*)d_in[17];
  float* out = (float*)d_out;

  // ---- workspace layout ----
  char* ws = (char*)d_ws;
  size_t off = 0;
  auto alloc = [&](size_t bytes) { size_t o = off; off = (off + bytes + 255) & ~(size_t)255; return o; };
  const size_t DD   = (size_t)Dm * Dm;
  const size_t BTD  = (size_t)BT * Dm;
  const size_t BTDc = (size_t)BT * DECC;

  size_t o_wq  = alloc(DD * 2), o_wk = alloc(DD * 2), o_wv = alloc(DD * 2);
  size_t o_wg  = alloc(DD * 2), o_wo = alloc(DD * 2);
  size_t o_d1T = alloc((size_t)Dm * DECC * 2);
  size_t o_d2T = alloc((size_t)Dm * DECC * 2);
  size_t o_xr  = alloc(BTD * 2);        // later reused as ybg (bf16)
  size_t o_xk  = alloc(BTD * 2);        // later reused (with o_xv) as y (f32)
  size_t o_xv  = alloc(BTD * 2);
  size_t o_xw  = alloc(BTD * 2);
  size_t o_xg  = alloc(BTD * 2);
  size_t o_q   = alloc(BTD * 4);
  size_t o_k   = alloc(BTD * 4);
  size_t o_v   = alloc(BTD * 4);
  size_t o_g   = alloc(BTD * 4);        // gpre -> g in place
  size_t o_hd1 = alloc(BTDc * 4);
  size_t o_th1 = alloc(BTDc * 2);
  size_t o_wd  = alloc(BTD * 4);        // wdelta -> logw in place

  bf16* wq_b  = (bf16*)(ws + o_wq);
  bf16* wk_b  = (bf16*)(ws + o_wk);
  bf16* wv_b  = (bf16*)(ws + o_wv);
  bf16* wg_b  = (bf16*)(ws + o_wg);
  bf16* wo_b  = (bf16*)(ws + o_wo);
  bf16* d1T_b = (bf16*)(ws + o_d1T);
  bf16* d2T_b = (bf16*)(ws + o_d2T);
  bf16* xr_b  = (bf16*)(ws + o_xr);
  bf16* xk_b  = (bf16*)(ws + o_xk);
  bf16* xv_b  = (bf16*)(ws + o_xv);
  bf16* xw_b  = (bf16*)(ws + o_xw);
  bf16* xg_b  = (bf16*)(ws + o_xg);
  float* qf   = (float*)(ws + o_q);
  float* kf   = (float*)(ws + o_k);
  float* vf   = (float*)(ws + o_v);
  float* gf   = (float*)(ws + o_g);
  float* hd1  = (float*)(ws + o_hd1);
  bf16* th1_b = (bf16*)(ws + o_th1);
  float* wd   = (float*)(ws + o_wd);
  // aliases (lifetimes disjoint from originals)
  float* yatt = (float*)(ws + o_xk);    // BTD f32 over xk_b+xv_b
  bf16* ybg_b = (bf16*)(ws + o_xr);     // BTD bf16 over xr_b

  const int TB = 256;
  // 1) weight conversions
  int nW = (int)DD;
  cvt_f2b_kernel<<<(nW + TB - 1) / TB, TB, 0, stream>>>(Wq, wq_b, nW);
  cvt_f2b_kernel<<<(nW + TB - 1) / TB, TB, 0, stream>>>(Wk, wk_b, nW);
  cvt_f2b_kernel<<<(nW + TB - 1) / TB, TB, 0, stream>>>(Wv, wv_b, nW);
  cvt_f2b_kernel<<<(nW + TB - 1) / TB, TB, 0, stream>>>(Wg, wg_b, nW);
  cvt_f2b_kernel<<<(nW + TB - 1) / TB, TB, 0, stream>>>(Wo, wo_b, nW);
  int nD = Dm * DECC;
  cvt_dw1T_kernel<<<(nD + TB - 1) / TB, TB, 0, stream>>>(decay_w1, d1T_b);
  cvt_dw2T_kernel<<<(nD + TB - 1) / TB, TB, 0, stream>>>(decay_w2, d2T_b);

  // 2) token-shift + mixing
  mix_kernel<<<BT, 256, 0, stream>>>(x, maa_x, maa_r, maa_k, maa_v, maa_w, maa_g,
                                     maa_w1, maa_w2, xr_b, xk_b, xv_b, xw_b, xg_b);

  // 3) projections (M=8192, N=2048, K=2048)
  dim3 gBig(Dm / 64, BT / 128);
  gemm_bf16_kernel<<<gBig, 256, 0, stream>>>(xr_b, wq_b, qf, BT, Dm, Dm);
  gemm_bf16_kernel<<<gBig, 256, 0, stream>>>(xk_b, wk_b, kf, BT, Dm, Dm);
  gemm_bf16_kernel<<<gBig, 256, 0, stream>>>(xv_b, wv_b, vf, BT, Dm, Dm);
  gemm_bf16_kernel<<<gBig, 256, 0, stream>>>(xg_b, wg_b, gf, BT, Dm, Dm);

  // 4) decay path: hd1 = xw @ decay_w1 ; th1 = tanh(hd1) ; wd = th1 @ decay_w2
  dim3 gDec(DECC / 64, BT / 128);
  gemm_bf16_kernel<<<gDec, 256, 0, stream>>>(xw_b, d1T_b, hd1, BT, DECC, Dm);
  int nH = BT * DECC;
  tanh_cvt_kernel<<<(nH + TB - 1) / TB, TB, 0, stream>>>(hd1, th1_b, nH);
  gemm_bf16_kernel<<<gBig, 256, 0, stream>>>(th1_b, d2T_b, wd, BT, Dm, DECC);

  // 5) logw, k-scaling, gate
  int nE = (int)BTD;
  prep_kernel<<<(nE + TB - 1) / TB, TB, 0, stream>>>(wd, kf, gf, time_decay, bg, nE);

  // 6) chunked linear attention (state resident in LDS)
  rwkv_attn_kernel<<<Bsz * NH, 128, 0, stream>>>(qf, kf, vf, wd, yatt);

  // 7) gate multiply + output projection
  mulg_kernel<<<(nE + TB - 1) / TB, TB, 0, stream>>>(yatt, gf, ybg_b, nE);
  gemm_bf16_kernel<<<gBig, 256, 0, stream>>>(ybg_b, wo_b, out, BT, Dm, Dm);
}
